// StructureVGAE_54030688584379
// MI455X (gfx1250) — matrix operand order repaired and checked by reading, not compiled, not run on hardware
//
#include <hip/hip_runtime.h>
#include <hip/hip_bf16.h>

#define N_NODES 10000
#define N_EDGES 320000
#define IN_C    512
#define HID     256
#define LAT     64
#define NT      5   // N-tiles per wave in the syrk kernel (625 = 5*125, exact)

typedef __attribute__((ext_vector_type(16))) __bf16 v16bf;
typedef __attribute__((ext_vector_type(8)))  __bf16 v8bf;
typedef __attribute__((ext_vector_type(8)))  float  v8f;

// ---------------------------------------------------------------------------
// f32-input fragment loaders (used by the three small feature GEMMs).
// A fragment: 16x32 bf16 (M x K), ISA 7.12.2 layout.
// ---------------------------------------------------------------------------
__device__ __forceinline__ v16bf load_a_frag(const float* __restrict__ Abase,
                                             int lda, int k0) {
  const int lane = threadIdx.x & 31;
  const int half = lane >> 4;
  const int r    = lane & 15;
  const float* ap = Abase + (size_t)r * lda + k0;
  v16bf a;
#pragma unroll
  for (int j = 0; j < 8; ++j) {
    const int kb = ((j & 4) << 2) + (half << 3) + ((j & 3) << 1);
    a[2 * j]     = (__bf16)ap[kb];
    a[2 * j + 1] = (__bf16)ap[kb + 1];
  }
  return a;
}

// B fragment: 32x16 bf16 (K x N), weights row-major KxN: B[k*ldb + n].
__device__ __forceinline__ v16bf load_b_frag(const float* __restrict__ Bbase,
                                             int ldb, int k0) {
  const int lane = threadIdx.x & 31;
  const int half = lane >> 4;
  const int n    = lane & 15;
  v16bf b;
#pragma unroll
  for (int j = 0; j < 8; ++j) {
    const int k = k0 + ((j & 4) << 2) + (half << 3) + ((j & 3) << 1);
    const float* bp = Bbase + (size_t)k * ldb + n;
    b[2 * j]     = (__bf16)bp[0];
    b[2 * j + 1] = (__bf16)bp[ldb];
  }
  return b;
}

// ---------------------------------------------------------------------------
// Feature GEMM: one wave per 16x16 tile, C = A(MxK) * B(KxN), bf16 WMMA.
// ---------------------------------------------------------------------------
__global__ void gemm_bf16_wmma(const float* __restrict__ A,
                               const float* __restrict__ B,
                               float* __restrict__ C,
                               int M, int Nc, int K, int lda, int ldb, int ldc) {
  const int wave    = blockIdx.x * (blockDim.x >> 5) + (threadIdx.x >> 5);
  const int tiles_n = Nc >> 4;
  const int tiles_m = M >> 4;
  if (wave >= tiles_m * tiles_n) return;  // uniform per-wave: EXEC stays all-1s

  const int tm = wave / tiles_n;
  const int tn = wave - tm * tiles_n;

  const float* Abase = A + (size_t)tm * 16 * lda;
  const float* Bbase = B + tn * 16;

  v8f c = {};
  for (int k0 = 0; k0 < K; k0 += 32) {
    if (k0 + 32 < K) {
      __builtin_prefetch(Abase + (size_t)(threadIdx.x & 15) * lda + k0 + 32, 0, 1);
    }
    v16bf a = load_a_frag(Abase, lda, k0);
    v16bf b = load_b_frag(Bbase, ldb, k0);
    c = __builtin_amdgcn_wmma_f32_16x16x32_bf16(false, a, false, b,
                                                (short)0, c, false, false);
  }

  const int lane = threadIdx.x & 31;
  const int half = lane >> 4;
  const int n    = lane & 15;
  float* Cp = C + (size_t)(tm * 16 + half * 8) * ldc + tn * 16 + n;
#pragma unroll
  for (int v = 0; v < 8; ++v) Cp[(size_t)v * ldc] = c[v];
}

// ---------------------------------------------------------------------------
// adj_pred = sigmoid(Z Z^T) from bf16 Z [N, 64] row-major.
// All fragment loads are contiguous 16B spans -> pure global_load_b128,
// zero conversion in the hot path. Each wave does a 16 x (16*NT) C slab,
// holding the two A fragments (K=64) in registers across NT sub-tiles.
// ---------------------------------------------------------------------------
__device__ __forceinline__ v16bf cat8(v8bf lo, v8bf hi) {
  v16bf r;
#pragma unroll
  for (int i = 0; i < 8; ++i) { r[i] = lo[i]; r[8 + i] = hi[i]; }
  return r;
}

// A fragment rows row0..row0+15, K slab k0..k0+31 (lane needs two 16B spans)
__device__ __forceinline__ v16bf load_a_bf(const __bf16* __restrict__ Z,
                                           int row0, int k0) {
  const int lane = threadIdx.x & 31;
  const int half = lane >> 4;
  const int r    = lane & 15;
  const __bf16* p = Z + (size_t)(row0 + r) * LAT + k0 + half * 16;
  return cat8(*(const v8bf*)p, *(const v8bf*)(p + 8));
}

// B^T fragment: B(k,n) = Z[col0+n][k]; lane needs one 32B span (two b128)
__device__ __forceinline__ v16bf load_bt_bf(const __bf16* __restrict__ Z,
                                            int col0, int k0) {
  const int lane = threadIdx.x & 31;
  const int half = lane >> 4;
  const int n    = lane & 15;
  const __bf16* p = Z + (size_t)(col0 + n) * LAT + k0 + half * 16;
  return cat8(*(const v8bf*)p, *(const v8bf*)(p + 8));
}

__global__ void syrk_sigmoid_wmma(const __bf16* __restrict__ Z,
                                  float* __restrict__ C) {
  const int tiles_m  = N_NODES / 16;        // 625
  const int groups_n = tiles_m / NT;        // 125
  const int wave = blockIdx.x * (blockDim.x >> 5) + (threadIdx.x >> 5);
  if (wave >= tiles_m * groups_n) return;

  const int tm   = wave / groups_n;
  const int tg   = wave - tm * groups_n;
  const int row0 = tm * 16;
  const int col0 = tg * (16 * NT);

  const v16bf a0 = load_a_bf(Z, row0, 0);
  const v16bf a1 = load_a_bf(Z, row0, 32);

  const v8f zero = {};
  v8f acc[NT];
#pragma unroll
  for (int t = 0; t < NT; ++t) {
    v16bf b0 = load_bt_bf(Z, col0 + t * 16, 0);
    v16bf b1 = load_bt_bf(Z, col0 + t * 16, 32);
    v8f c = zero;
    c = __builtin_amdgcn_wmma_f32_16x16x32_bf16(false, a0, false, b0,
                                                (short)0, c, false, false);
    c = __builtin_amdgcn_wmma_f32_16x16x32_bf16(false, a1, false, b1,
                                                (short)0, c, false, false);
    acc[t] = c;
  }

  const int lane = threadIdx.x & 31;
  const int half = lane >> 4;
  const int n    = lane & 15;
#pragma unroll
  for (int t = 0; t < NT; ++t) {
    float* Cp = C + (size_t)(row0 + half * 8) * N_NODES + col0 + t * 16 + n;
#pragma unroll
    for (int v = 0; v < 8; ++v) {
      Cp[(size_t)v * N_NODES] = 1.0f / (1.0f + __expf(-acc[t][v]));
    }
  }
}

// ---------------------------------------------------------------------------
// GCN scatter machinery
// ---------------------------------------------------------------------------
__global__ void fill_f32(float* __restrict__ p, float v, int n) {
  int i = blockIdx.x * blockDim.x + threadIdx.x;
  if (i < n) p[i] = v;
}

__global__ void deg_count(const long long* __restrict__ dst, float* __restrict__ deg, int E) {
  int i = blockIdx.x * blockDim.x + threadIdx.x;
  if (i < E) atomicAdd(&deg[(int)dst[i]], 1.0f);
}

__global__ void rsqrt_deg(const float* __restrict__ deg, float* __restrict__ dinv, int n) {
  int i = blockIdx.x * blockDim.x + threadIdx.x;
  if (i < n) dinv[i] = rsqrtf(fmaxf(deg[i], 1e-12f));
}

__global__ void bias_init(float* __restrict__ out, const float* __restrict__ b,
                          int n, int mask) {
  int i = blockIdx.x * blockDim.x + threadIdx.x;
  if (i < n) out[i] = b[i & mask];
}

__global__ void scatter_edges(const float* __restrict__ h,
                              const long long* __restrict__ src,
                              const long long* __restrict__ dst,
                              const float* __restrict__ dinv,
                              float* __restrict__ out, int F) {
  const int e = blockIdx.x;
  const int s = (int)src[e];
  const int d = (int)dst[e];
  const float norm = dinv[s] * dinv[d];
  const int f = threadIdx.x;
  atomicAdd(&out[(size_t)d * F + f], h[(size_t)s * F + f] * norm);
}

__global__ void selfloop_act(const float* __restrict__ h,
                             const float* __restrict__ dinv,
                             float* __restrict__ out, int F, int relu) {
  const int i = blockIdx.x;
  const int f = threadIdx.x;
  const float di = dinv[i];
  const size_t idx = (size_t)i * F + f;
  float v = out[idx] + h[idx] * di * di;
  if (relu) v = fmaxf(v, 0.0f);
  out[idx] = v;
}

__global__ void copy_f32(const float* __restrict__ a, float* __restrict__ b, int n) {
  int i = blockIdx.x * blockDim.x + threadIdx.x;
  if (i < n) b[i] = a[i];
}

__global__ void cvt_f32_bf16(const float* __restrict__ a, __bf16* __restrict__ b, int n) {
  int i = blockIdx.x * blockDim.x + threadIdx.x;
  if (i < n) b[i] = (__bf16)a[i];
}

// ---------------------------------------------------------------------------
extern "C" void kernel_launch(void* const* d_in, const int* in_sizes, int n_in,
                              void* d_out, int out_size, void* d_ws, size_t ws_size,
                              hipStream_t stream) {
  const float*     x   = (const float*)d_in[0];
  const long long* ei  = (const long long*)d_in[1];  // int64 [2,E]
  const float*     W1  = (const float*)d_in[2];
  const float*     b1  = (const float*)d_in[3];
  const float*     Wmu = (const float*)d_in[4];
  const float*     bmu = (const float*)d_in[5];
  const float*     Wlv = (const float*)d_in[6];
  const float*     blv = (const float*)d_in[7];

  const long long* src = ei;
  const long long* dst = ei + N_EDGES;

  // d_out: mu | logvar | z | adj_pred
  float* out    = (float*)d_out;
  float* mu_out = out;
  float* lv_out = out + (size_t)N_NODES * LAT;
  float* z_out  = out + (size_t)2 * N_NODES * LAT;
  float* adj    = out + (size_t)3 * N_NODES * LAT;

  // workspace layout (~27 MB)
  float* ws   = (float*)d_ws;
  float* deg  = ws;                            // N
  float* dinv = ws + 16384;                    // N
  float* H0   = ws + 32768;                    // N*HID
  float* H    = H0 + (size_t)N_NODES * HID;    // N*HID
  float* MuP  = H  + (size_t)N_NODES * HID;    // N*LAT
  float* LvP  = MuP + (size_t)N_NODES * LAT;   // N*LAT
  __bf16* zbf = (__bf16*)(LvP + (size_t)N_NODES * LAT);  // N*LAT bf16 (16B aligned)

  // ---- degree / norm -------------------------------------------------------
  fill_f32<<<(N_NODES + 255) / 256, 256, 0, stream>>>(deg, 1.0f, N_NODES);
  deg_count<<<(N_EDGES + 255) / 256, 256, 0, stream>>>(dst, deg, N_EDGES);
  rsqrt_deg<<<(N_NODES + 255) / 256, 256, 0, stream>>>(deg, dinv, N_NODES);

  // ---- conv1: H = relu(scatter(x@W1) + b1) ---------------------------------
  {
    int tiles  = (N_NODES / 16) * (HID / 16);
    gemm_bf16_wmma<<<(tiles + 7) / 8, 256, 0, stream>>>(
        x, W1, H0, N_NODES, HID, IN_C, IN_C, HID, HID);
  }
  bias_init<<<(N_NODES * HID + 255) / 256, 256, 0, stream>>>(H, b1, N_NODES * HID, HID - 1);
  scatter_edges<<<N_EDGES, HID, 0, stream>>>(H0, src, dst, dinv, H, HID);
  selfloop_act<<<N_NODES, HID, 0, stream>>>(H0, dinv, H, HID, 1);

  // ---- conv-mu -------------------------------------------------------------
  {
    int tiles  = (N_NODES / 16) * (LAT / 16);
    gemm_bf16_wmma<<<(tiles + 7) / 8, 256, 0, stream>>>(
        H, Wmu, MuP, N_NODES, LAT, HID, HID, LAT, LAT);
  }
  bias_init<<<(N_NODES * LAT + 255) / 256, 256, 0, stream>>>(mu_out, bmu, N_NODES * LAT, LAT - 1);
  scatter_edges<<<N_EDGES, LAT, 0, stream>>>(MuP, src, dst, dinv, mu_out, LAT);
  selfloop_act<<<N_NODES, LAT, 0, stream>>>(MuP, dinv, mu_out, LAT, 0);

  // ---- conv-logvar ---------------------------------------------------------
  {
    int tiles  = (N_NODES / 16) * (LAT / 16);
    gemm_bf16_wmma<<<(tiles + 7) / 8, 256, 0, stream>>>(
        H, Wlv, LvP, N_NODES, LAT, HID, HID, LAT, LAT);
  }
  bias_init<<<(N_NODES * LAT + 255) / 256, 256, 0, stream>>>(lv_out, blv, N_NODES * LAT, LAT - 1);
  scatter_edges<<<N_EDGES, LAT, 0, stream>>>(LvP, src, dst, dinv, lv_out, LAT);
  selfloop_act<<<N_NODES, LAT, 0, stream>>>(LvP, dinv, lv_out, LAT, 0);

  // ---- z = mu; zbf = bf16(z) ----------------------------------------------
  copy_f32<<<(N_NODES * LAT + 255) / 256, 256, 0, stream>>>(mu_out, z_out, N_NODES * LAT);
  cvt_f32_bf16<<<(N_NODES * LAT + 255) / 256, 256, 0, stream>>>(z_out, zbf, N_NODES * LAT);

  // ---- adj_pred = sigmoid(z @ z^T) -----------------------------------------
  {
    int tiles = (N_NODES / 16) * ((N_NODES / 16) / NT);  // 625 * 125
    syrk_sigmoid_wmma<<<(tiles + 7) / 8, 256, 0, stream>>>(zbf, adj);
  }
}
